// W4A16QMamba_40243843564258
// MI455X (gfx1250) — compile-verified, matrix operand order checked by reading, not run on hardware
//
#include <hip/hip_runtime.h>
#include <hip/hip_bf16.h>
#include <math.h>

// ---------------------------------------------------------------------------
// Types for WMMA fragments (gfx1250, wave32)
// ---------------------------------------------------------------------------
typedef __bf16 bf16_t;
typedef __attribute__((ext_vector_type(16))) __bf16 v16bf;
typedef __attribute__((ext_vector_type(8)))  __bf16 v8bf;
typedef __attribute__((ext_vector_type(8)))  float  v8f;

#define D_MODEL 2048
#define D_STATE 16
#define D_CONV  4
#define D_INNER 4096
#define DT_RANK 128
#define BATCH   2
#define SEQLEN  1024
#define NROWS   (BATCH * SEQLEN)          // 2048 token rows
#define XPROJ_N (DT_RANK + 2 * D_STATE)   // 160

#define SHUF16(lo, hi) __builtin_shufflevector(lo, hi, 0, 1, 2, 3, 4, 5, 6, 7, \
                                               8, 9, 10, 11, 12, 13, 14, 15)

// ---------------------------------------------------------------------------
// Generic f32 -> bf16 conversion (handles strided sub-matrix extraction)
// ---------------------------------------------------------------------------
__global__ void k_cvt_bf16(const float* __restrict__ src, bf16_t* __restrict__ dst,
                           unsigned long long rows, unsigned int cols,
                           unsigned int srcStride) {
    unsigned long long gid = (unsigned long long)blockIdx.x * blockDim.x + threadIdx.x;
    unsigned long long n = rows * (unsigned long long)cols;
    if (gid >= n) return;
    unsigned long long r = gid / cols;
    unsigned int c = (unsigned int)(gid - r * cols);
    dst[gid] = (bf16_t)src[r * (unsigned long long)srcStride + c];
}

// ---------------------------------------------------------------------------
// LDS-staged, double-buffered bf16 WMMA GEMM.
//   C[M x N] (f32, row-major) = A[M x K] * B[N x K]^T,  A/B bf16 row-major.
// Block: 256 threads (8 waves), block tile 128x128, wave sub-tile 32x64
// (2 M-tiles x 4 N-tiles = 8 WMMA accumulators/wave).
// LDS row stride padded to 40 elems (80B = 20 banks): the 16 rows touched by a
// fragment load land in distinct banks, and every ds_* stays 16B aligned.
// Requires M%128==0, N%128==0, K%32==0.
// ---------------------------------------------------------------------------
__global__ __launch_bounds__(256) void k_gemm_bf16_wmma_lds(
        const bf16_t* __restrict__ A, const bf16_t* __restrict__ B,
        float* __restrict__ C, int M, int N, int K) {
    constexpr int LSTR = 40;                 // padded LDS row stride (elems)
    constexpr int TILE = 128 * LSTR;         // 5120 elems per tile buffer
    __shared__ __align__(16) bf16_t smem[4 * TILE];   // A[2] then B[2]
    bf16_t* __restrict__ ldsA = smem;
    bf16_t* __restrict__ ldsB = smem + 2 * TILE;

    const int t    = (int)threadIdx.x;
    const int lane = t & 31;
    const int w    = t >> 5;                 // wave 0..7
    const int wm   = w & 3;                  // 4 waves along M (32 rows each)
    const int wn   = w >> 2;                 // 2 waves along N (64 cols each)

    const int nTiles = N >> 7;
    const int bm = (int)(blockIdx.x / nTiles) << 7;
    const int bn = (int)(blockIdx.x % nTiles) << 7;

    const int half = (lane >> 4) << 3;       // 0 or 8: K sub-half per lane group
    const int mrow = lane & 15;

    // Global/LDS chunk coordinates: 512 16B-chunks per 128x32 tile, 2/thread.
    const int c0 = t, c1 = t + 256;
    const int r0 = c0 >> 2, o0 = (c0 & 3) << 3;
    const int r1 = c1 >> 2, o1 = (c1 & 3) << 3;

    v8f acc[2][4];
    #pragma unroll
    for (int mi = 0; mi < 2; ++mi)
        #pragma unroll
        for (int ni = 0; ni < 4; ++ni)
            acc[mi][ni] = (v8f){};

    const int nk = K >> 5;

    // ---- prologue: tile 0 -> buffer 0 ----
    uint4 rA0 = *(const uint4*)(A + (size_t)(bm + r0) * K + o0);
    uint4 rA1 = *(const uint4*)(A + (size_t)(bm + r1) * K + o1);
    uint4 rB0 = *(const uint4*)(B + (size_t)(bn + r0) * K + o0);
    uint4 rB1 = *(const uint4*)(B + (size_t)(bn + r1) * K + o1);
    *(uint4*)(ldsA + r0 * LSTR + o0) = rA0;
    *(uint4*)(ldsA + r1 * LSTR + o1) = rA1;
    *(uint4*)(ldsB + r0 * LSTR + o0) = rB0;
    *(uint4*)(ldsB + r1 * LSTR + o1) = rB1;
    __syncthreads();

    for (int kt = 0; kt < nk; ++kt) {
        const int cur = kt & 1;
        // Issue next tile's global loads early; they drain while we compute.
        if (kt + 1 < nk) {
            const int k0 = (kt + 1) << 5;
            rA0 = *(const uint4*)(A + (size_t)(bm + r0) * K + k0 + o0);
            rA1 = *(const uint4*)(A + (size_t)(bm + r1) * K + k0 + o1);
            rB0 = *(const uint4*)(B + (size_t)(bn + r0) * K + k0 + o0);
            rB1 = *(const uint4*)(B + (size_t)(bn + r1) * K + k0 + o1);
        }

        const bf16_t* __restrict__ As = ldsA + cur * TILE;
        const bf16_t* __restrict__ Bs = ldsB + cur * TILE;

        v16bf af[2], bfr[4];
        #pragma unroll
        for (int mi = 0; mi < 2; ++mi) {
            const int r = wm * 32 + mi * 16 + mrow;
            v8bf lo = *(const v8bf*)(As + r * LSTR + half);
            v8bf hi = *(const v8bf*)(As + r * LSTR + 16 + half);
            af[mi] = SHUF16(lo, hi);
        }
        #pragma unroll
        for (int ni = 0; ni < 4; ++ni) {
            const int r = wn * 64 + ni * 16 + mrow;
            v8bf lo = *(const v8bf*)(Bs + r * LSTR + half);
            v8bf hi = *(const v8bf*)(Bs + r * LSTR + 16 + half);
            bfr[ni] = SHUF16(lo, hi);
        }

        #pragma unroll
        for (int mi = 0; mi < 2; ++mi)
            #pragma unroll
            for (int ni = 0; ni < 4; ++ni)
                acc[mi][ni] = __builtin_amdgcn_wmma_f32_16x16x32_bf16(
                    false, af[mi], false, bfr[ni], (short)0, acc[mi][ni],
                    false, false);

        if (kt + 1 < nk) {
            __syncthreads();                 // everyone done reading nxt buffer
            const int nxt = cur ^ 1;
            bf16_t* __restrict__ Ad = ldsA + nxt * TILE;
            bf16_t* __restrict__ Bd = ldsB + nxt * TILE;
            *(uint4*)(Ad + r0 * LSTR + o0) = rA0;
            *(uint4*)(Ad + r1 * LSTR + o1) = rA1;
            *(uint4*)(Bd + r0 * LSTR + o0) = rB0;
            *(uint4*)(Bd + r1 * LSTR + o1) = rB1;
            __syncthreads();
        }
    }

    // ---- epilogue: C/D layout: lane<16 -> rows +0..7, lane>=16 -> rows +8..15
    const int rwave = bm + wm * 32 + ((lane >> 4) << 3);
    const int ccol  = lane & 15;
    #pragma unroll
    for (int mi = 0; mi < 2; ++mi) {
        #pragma unroll
        for (int ni = 0; ni < 4; ++ni) {
            const int rb = rwave + mi * 16;
            const int cb = bn + wn * 64 + ni * 16 + ccol;
            #pragma unroll
            for (int r = 0; r < 8; ++r)
                C[(size_t)(rb + r) * N + cb] = acc[mi][ni][r];
        }
    }
}

// ---------------------------------------------------------------------------
// Direct-from-global WMMA GEMM for small N (x_proj, N=160).
// One wave -> 16x32 output tile. Requires M%16==0, N%32==0, K%32==0.
// ---------------------------------------------------------------------------
__global__ void k_gemm_bf16_wmma(const bf16_t* __restrict__ A,
                                 const bf16_t* __restrict__ B,
                                 float* __restrict__ C,
                                 int M, int N, int K) {
    const int wave = (int)((blockIdx.x * blockDim.x + threadIdx.x) >> 5);
    const int lane = (int)(threadIdx.x & 31);
    const int tilesN2 = N >> 5;
    const int tm = wave / tilesN2;              // wave-uniform
    const int tn = wave - tm * tilesN2;
    if (tm >= (M >> 4)) return;                 // whole-wave exit

    const int m0 = tm << 4;
    const int n0 = tn << 5;
    const int half = (lane >> 4) << 3;

    const bf16_t* __restrict__ Arow  = A + (size_t)(m0 + (lane & 15)) * K;
    const bf16_t* __restrict__ Brow0 = B + (size_t)(n0 + (lane & 15)) * K;
    const bf16_t* __restrict__ Brow1 = Brow0 + (size_t)16 * K;

    v8f acc0 = {};
    v8f acc1 = {};

    for (int k0 = 0; k0 < K; k0 += 32) {
        if (k0 + 64 <= K) {
            __builtin_prefetch(Arow  + k0 + 32, 0, 1);
            __builtin_prefetch(Brow0 + k0 + 32, 0, 1);
            __builtin_prefetch(Brow1 + k0 + 32, 0, 1);
        }
        v8bf alo = *(const v8bf*)(Arow + k0 + half);
        v8bf ahi = *(const v8bf*)(Arow + k0 + 16 + half);
        v16bf a = SHUF16(alo, ahi);
        v8bf b0lo = *(const v8bf*)(Brow0 + k0 + half);
        v8bf b0hi = *(const v8bf*)(Brow0 + k0 + 16 + half);
        v16bf b0 = SHUF16(b0lo, b0hi);
        v8bf b1lo = *(const v8bf*)(Brow1 + k0 + half);
        v8bf b1hi = *(const v8bf*)(Brow1 + k0 + 16 + half);
        v16bf b1 = SHUF16(b1lo, b1hi);

        acc0 = __builtin_amdgcn_wmma_f32_16x16x32_bf16(
            false, a, false, b0, (short)0, acc0, false, false);
        acc1 = __builtin_amdgcn_wmma_f32_16x16x32_bf16(
            false, a, false, b1, (short)0, acc1, false, false);
    }

    const int crow = m0 + ((lane >> 4) << 3);
    const int ccol = lane & 15;
    for (int r = 0; r < 8; ++r) {
        C[(size_t)(crow + r) * N + n0 + ccol]      = acc0[r];
        C[(size_t)(crow + r) * N + n0 + 16 + ccol] = acc1[r];
    }
}

// ---------------------------------------------------------------------------
// Causal depthwise conv1d + bias + SiLU over x half of xz.
// ---------------------------------------------------------------------------
__global__ void k_conv_silu(const float* __restrict__ xz,
                            const float* __restrict__ cw,
                            const float* __restrict__ cb,
                            float* __restrict__ xf, bf16_t* __restrict__ xh) {
    unsigned long long gid = (unsigned long long)blockIdx.x * blockDim.x + threadIdx.x;
    if (gid >= (unsigned long long)NROWS * D_INNER) return;
    const int d = (int)(gid & (D_INNER - 1));
    const unsigned long long row = gid >> 12;           // b*L + l
    const int l = (int)(row & (SEQLEN - 1));
    const unsigned long long rowBase = row - (unsigned long long)l;

    float acc = cb[d];
    #pragma unroll
    for (int t = 0; t < D_CONV; ++t) {
        const int ls = l - (D_CONV - 1) + t;
        if (ls >= 0)
            acc += xz[(rowBase + ls) * (2ull * D_INNER) + d] * cw[d * D_CONV + t];
    }
    const float s = acc / (1.0f + __expf(-acc));        // SiLU
    xf[gid] = s;
    xh[gid] = (bf16_t)s;
}

// ---------------------------------------------------------------------------
// delta = softplus(delta + dt_bias[d])
// ---------------------------------------------------------------------------
__global__ void k_softplus_bias(float* __restrict__ delta,
                                const float* __restrict__ dt_bias) {
    unsigned long long gid = (unsigned long long)blockIdx.x * blockDim.x + threadIdx.x;
    if (gid >= (unsigned long long)NROWS * D_INNER) return;
    const int d = (int)(gid & (D_INNER - 1));
    const float v = delta[gid] + dt_bias[d];
    delta[gid] = (v > 20.0f) ? v : log1pf(__expf(v));
}

// ---------------------------------------------------------------------------
// Selective scan: one thread per (b, d) channel; n=16 state in registers.
// ---------------------------------------------------------------------------
__global__ void k_scan(const float* __restrict__ delta,
                       const float* __restrict__ u,
                       const float* __restrict__ xdbl,   // (b*L, 160)
                       const float* __restrict__ xz,     // z at col D_INNER+d
                       const float* __restrict__ A_log,
                       const float* __restrict__ Dp,
                       float* __restrict__ y) {
    const int gid = (int)(blockIdx.x * blockDim.x + threadIdx.x);
    if (gid >= BATCH * D_INNER) return;
    const int d = gid & (D_INNER - 1);
    const int b = gid >> 12;

    float Ac[D_STATE];
    #pragma unroll
    for (int j = 0; j < D_STATE; ++j)
        Ac[j] = -__expf(A_log[d * D_STATE + j]);
    const float Dd = Dp[d];

    float st[D_STATE];
    #pragma unroll
    for (int j = 0; j < D_STATE; ++j) st[j] = 0.0f;

    for (int l = 0; l < SEQLEN; ++l) {
        const unsigned long long row = (unsigned long long)b * SEQLEN + l;
        const float dt = delta[row * D_INNER + d];
        const float uu = u[row * D_INNER + d];
        const float dtu = dt * uu;
        const float* __restrict__ xd = xdbl + row * XPROJ_N;
        float yv = 0.0f;
        #pragma unroll
        for (int j = 0; j < D_STATE; ++j) {
            const float dA = __expf(dt * Ac[j]);
            st[j] = st[j] * dA + dtu * xd[DT_RANK + j];            // B
            yv += st[j] * xd[DT_RANK + D_STATE + j];               // C
        }
        yv += Dd * uu;
        const float z = xz[row * (2ull * D_INNER) + D_INNER + d];
        const float g = z / (1.0f + __expf(-z));                   // SiLU gate
        y[row * D_INNER + d] = yv * g;
    }
}

// ---------------------------------------------------------------------------
// Fast Walsh–Hadamard transform per row (4096 elems, 12 LDS stages),
// scaled by 1/sqrt(4096), output bf16 for the final WMMA GEMM.
// ---------------------------------------------------------------------------
__global__ void k_fwht(const float* __restrict__ y, bf16_t* __restrict__ yh) {
    __shared__ float buf[D_INNER];
    const unsigned long long row = blockIdx.x;
    const float* __restrict__ src = y + row * D_INNER;
    for (int i = threadIdx.x; i < D_INNER; i += blockDim.x) buf[i] = src[i];
    __syncthreads();
    for (int len = 1; len < D_INNER; len <<= 1) {
        for (int idx = threadIdx.x; idx < (D_INNER >> 1); idx += blockDim.x) {
            const int i0 = ((idx & ~(len - 1)) << 1) | (idx & (len - 1));
            const int i1 = i0 + len;
            const float a = buf[i0];
            const float c = buf[i1];
            buf[i0] = a + c;
            buf[i1] = a - c;
        }
        __syncthreads();
    }
    bf16_t* __restrict__ dst = yh + row * D_INNER;
    const float scale = 1.0f / 64.0f;   // 1/sqrt(4096)
    for (int i = threadIdx.x; i < D_INNER; i += blockDim.x)
        dst[i] = (bf16_t)(buf[i] * scale);
}

// ---------------------------------------------------------------------------
// Host-side orchestration
// ---------------------------------------------------------------------------
static inline unsigned int div_up_u(unsigned long long n, unsigned int d) {
    return (unsigned int)((n + d - 1) / d);
}

extern "C" void kernel_launch(void* const* d_in, const int* in_sizes, int n_in,
                              void* d_out, int out_size, void* d_ws, size_t ws_size,
                              hipStream_t stream) {
    (void)in_sizes; (void)n_in; (void)out_size; (void)ws_size;
    const float* hidden  = (const float*)d_in[0];  // (2,1024,2048)
    const float* W_in    = (const float*)d_in[1];  // (8192,2048)
    const float* conv_w  = (const float*)d_in[2];  // (4096,4)
    const float* conv_b  = (const float*)d_in[3];  // (4096,)
    const float* W_x     = (const float*)d_in[4];  // (160,4096)
    const float* W_dt    = (const float*)d_in[5];  // (4096,128)
    const float* dt_bias = (const float*)d_in[6];  // (4096,)
    const float* A_log   = (const float*)d_in[7];  // (4096,16)
    const float* Dp      = (const float*)d_in[8];  // (4096,)
    const float* W_out   = (const float*)d_in[9];  // (2048,4096)
    float* out = (float*)d_out;                    // (2,1024,2048) f32

    // ---- workspace carve-up (256B aligned) ----
    char* ws = (char*)d_ws;
    size_t off = 0;
    auto alloc = [&](size_t bytes) -> char* {
        char* p = ws + off;
        off += (bytes + 255) & ~(size_t)255;
        return p;
    };
    bf16_t* w_in_h  = (bf16_t*)alloc((size_t)8192 * 2048 * 2);
    bf16_t* w_out_h = (bf16_t*)alloc((size_t)2048 * 4096 * 2);
    bf16_t* w_x_h   = (bf16_t*)alloc((size_t)XPROJ_N * 4096 * 2);
    bf16_t* w_dt_h  = (bf16_t*)alloc((size_t)4096 * 128 * 2);
    bf16_t* hid_h   = (bf16_t*)alloc((size_t)NROWS * 2048 * 2);
    float*  xz_f    = (float*) alloc((size_t)NROWS * 8192 * 4);
    float*  xc_f    = (float*) alloc((size_t)NROWS * 4096 * 4);
    bf16_t* xc_h    = (bf16_t*)alloc((size_t)NROWS * 4096 * 2);
    float*  xdbl_f  = (float*) alloc((size_t)NROWS * XPROJ_N * 4);
    bf16_t* dtr_h   = (bf16_t*)alloc((size_t)NROWS * DT_RANK * 2);
    float*  delta_f = (float*) alloc((size_t)NROWS * 4096 * 4);
    float*  y_f     = (float*) alloc((size_t)NROWS * 4096 * 4);
    bf16_t* y_h     = (bf16_t*)alloc((size_t)NROWS * 4096 * 2);

    const dim3 B256(256);

    // ---- one-time bf16 conversions ----
    k_cvt_bf16<<<div_up_u((size_t)8192 * 2048, 256), B256, 0, stream>>>(
        W_in, w_in_h, 8192ull, 2048u, 2048u);
    k_cvt_bf16<<<div_up_u((size_t)2048 * 4096, 256), B256, 0, stream>>>(
        W_out, w_out_h, 2048ull, 4096u, 4096u);
    k_cvt_bf16<<<div_up_u((size_t)XPROJ_N * 4096, 256), B256, 0, stream>>>(
        W_x, w_x_h, (unsigned long long)XPROJ_N, 4096u, 4096u);
    k_cvt_bf16<<<div_up_u((size_t)4096 * 128, 256), B256, 0, stream>>>(
        W_dt, w_dt_h, 4096ull, 128u, 128u);
    k_cvt_bf16<<<div_up_u((size_t)NROWS * 2048, 256), B256, 0, stream>>>(
        hidden, hid_h, (unsigned long long)NROWS, 2048u, 2048u);

    // ---- in_proj: xz = hidden @ W_in^T  (M=2048, N=8192, K=2048) ----
    k_gemm_bf16_wmma_lds<<<(2048 / 128) * (8192 / 128), B256, 0, stream>>>(
        hid_h, w_in_h, xz_f, 2048, 8192, 2048);

    // ---- depthwise causal conv + SiLU ----
    k_conv_silu<<<div_up_u((size_t)NROWS * 4096, 256), B256, 0, stream>>>(
        xz_f, conv_w, conv_b, xc_f, xc_h);

    // ---- x_proj: x_dbl = x @ W_x^T  (M=2048, N=160, K=4096) ----
    {
        const int waves = (2048 / 16) * (XPROJ_N / 32);
        k_gemm_bf16_wmma<<<div_up_u((size_t)waves * 32, 256), B256, 0, stream>>>(
            xc_h, w_x_h, xdbl_f, 2048, XPROJ_N, 4096);
    }

    // ---- dt_proj: delta = dt_r @ W_dt^T  (M=2048, N=4096, K=128) ----
    k_cvt_bf16<<<div_up_u((size_t)NROWS * DT_RANK, 256), B256, 0, stream>>>(
        xdbl_f, dtr_h, (unsigned long long)NROWS, (unsigned)DT_RANK, (unsigned)XPROJ_N);
    k_gemm_bf16_wmma_lds<<<(2048 / 128) * (4096 / 128), B256, 0, stream>>>(
        dtr_h, w_dt_h, delta_f, 2048, 4096, 128);
    k_softplus_bias<<<div_up_u((size_t)NROWS * 4096, 256), B256, 0, stream>>>(
        delta_f, dt_bias);

    // ---- selective scan (+ D*u, SiLU(z) gate) ----
    k_scan<<<div_up_u((size_t)BATCH * D_INNER, 256), B256, 0, stream>>>(
        delta_f, xc_f, xdbl_f, xz_f, A_log, Dp, y_f);

    // ---- normalized Hadamard via FWHT ----
    k_fwht<<<NROWS, B256, 0, stream>>>(y_f, y_h);

    // ---- out_proj: out = yH @ W_out^T  (M=2048, N=2048, K=4096) ----
    k_gemm_bf16_wmma_lds<<<(2048 / 128) * (2048 / 128), B256, 0, stream>>>(
        y_h, w_out_h, out, 2048, 2048, 4096);
}
